// AttentionBlock_41798621724711
// MI455X (gfx1250) — matrix operand order repaired and checked by reading, not compiled, not run on hardware
//
#include <hip/hip_runtime.h>
#include <hip/hip_bf16.h>

// ---------------- problem constants ----------------
#define B_    64
#define CIMG  256
#define HW    4096
#define HID_  512
#define T_    128
#define M_    128
#define A_    128
#define NTOT  (HW + T_)   // 4224
#define STR   40          // padded LDS k-stride (bf16 units): bank-conflict-free

typedef __attribute__((ext_vector_type(16))) __bf16 v16bf;
typedef __attribute__((ext_vector_type(8)))  __bf16 v8bf;
typedef __attribute__((ext_vector_type(4)))  __bf16 v4bf;
typedef __attribute__((ext_vector_type(8)))  float  v8f;

// Branch-free tanh: v_exp_f32 + v_rcp_f32, saturates correctly at +/-1.
__device__ __forceinline__ float fast_tanh(float x) {
  float e = __expf(2.0f * x);
  return 1.0f - 2.0f * __builtin_amdgcn_rcpf(e + 1.0f);
}

// ---------------- WMMA fragment helpers (wave32, 16x16x32 bf16) ----------------
// A 16x32: lane row m = lane&15; element e: K = half*16 + e.
// With LDS layout sA[row*STR + k], both 8-element chunks are contiguous 16B.
__device__ __forceinline__ v16bf lds_load_a(const __bf16* __restrict__ row, int half) {
  v8bf lo = *(const v8bf*)(row + half * 16);
  v8bf hi = *(const v8bf*)(row + half * 16 + 8);
  v16bf a;
#pragma unroll
  for (int e = 0; e < 8; ++e) { a[e] = lo[e]; a[8 + e] = hi[e]; }
  return a;
}

// B 32x16: lane col n = lane&15; element e: K = half*16 + e -> contiguous 32B run.
__device__ __forceinline__ v16bf lds_load_b(const __bf16* __restrict__ col, int half) {
  v8bf lo = *(const v8bf*)(col + half * 16);
  v8bf hi = *(const v8bf*)(col + half * 16 + 8);
  v16bf b;
#pragma unroll
  for (int e = 0; e < 8; ++e) { b[e] = lo[e]; b[8 + e] = hi[e]; }
  return b;
}

// B fragment directly from global bf16 featT[n][m] (k = m contiguous).
__device__ __forceinline__ v16bf glb_load_b(const __bf16* __restrict__ col, int half) {
  v8bf lo = *(const v8bf*)(col + half * 16);
  v8bf hi = *(const v8bf*)(col + half * 16 + 8);
  v16bf b;
#pragma unroll
  for (int e = 0; e < 8; ++e) { b[e] = lo[e]; b[8 + e] = hi[e]; }
  return b;
}

__device__ __forceinline__ v8f wmma_bf16(v16bf a, v16bf b, v8f c) {
  return __builtin_amdgcn_wmma_f32_16x16x32_bf16(false, a, false, b, (short)0, c,
                                                 false, false);
}

// ---------- kernel 1/2: featT[b][n][m] = tanh(W[M,K] @ X[b][K,NLD]) + col-sum partials ----------
// block = (n-tile of 64, batch); 256 threads = 8 waves; wave w -> m-tile w, 4 n-subtiles.
template <int K, int NLD>
__global__ void k_ingemm(const float* __restrict__ W, const float* __restrict__ X,
                         __bf16* __restrict__ featT, float* __restrict__ sumP) {
  __shared__ __bf16 sA[M_ * STR];  // 10 KB: W slice [128 m][32 k]
  __shared__ __bf16 sB[64 * STR];  // 5 KB: X tile transposed [64 n][32 k]
  const int b = blockIdx.y;
  const int nblk = blockIdx.x * 64;
  const int tid = threadIdx.x;
  const int wave = tid >> 5, lane = tid & 31;
  const int n16 = lane & 15, half = lane >> 4;
  const int m0 = wave * 16;
  const float* Xb = X + (size_t)b * K * NLD;
  __bf16* Yb = featT + (size_t)b * NLD * M_;

  // staging decomposition
  const int kr = tid >> 3;          // 0..31  (X row within k-slice)
  const int nc = (tid & 7) * 8;     // 0..56  (X col group of 8)
  v8f acc[4] = {};

  for (int k0 = 0; k0 < K; k0 += 32) {
    __syncthreads();  // previous iteration's LDS reads complete
    // stage W slice [128 x 32] f32 -> bf16, layout sA[m*STR + kk]
#pragma unroll
    for (int it = 0; it < 4; ++it) {
      int s = tid * 4 + it * 1024;      // 0..4095
      int m = s >> 5, kk = s & 31;
      float4 w = *(const float4*)(W + (size_t)m * K + k0 + kk);
      v4bf p;
      p[0] = (__bf16)w.x; p[1] = (__bf16)w.y; p[2] = (__bf16)w.z; p[3] = (__bf16)w.w;
      *(v4bf*)(sA + m * STR + kk) = p;
    }
    // stage X tile [32 k x 64 n] f32 -> bf16 transposed sB[n*STR + k]
    {
      const float* src = Xb + (size_t)(k0 + kr) * NLD + nblk + nc;
      float4 x0 = *(const float4*)(src);
      float4 x1 = *(const float4*)(src + 4);
      sB[(nc + 0) * STR + kr] = (__bf16)x0.x;
      sB[(nc + 1) * STR + kr] = (__bf16)x0.y;
      sB[(nc + 2) * STR + kr] = (__bf16)x0.z;
      sB[(nc + 3) * STR + kr] = (__bf16)x0.w;
      sB[(nc + 4) * STR + kr] = (__bf16)x1.x;
      sB[(nc + 5) * STR + kr] = (__bf16)x1.y;
      sB[(nc + 6) * STR + kr] = (__bf16)x1.z;
      sB[(nc + 7) * STR + kr] = (__bf16)x1.w;
    }
    __syncthreads();
    v16bf a = lds_load_a(sA + (m0 + n16) * STR, half);
#pragma unroll
    for (int j = 0; j < 4; ++j) {
      v16bf bm = lds_load_b(sB + (j * 16 + n16) * STR, half);
      acc[j] = wmma_bf16(a, bm, acc[j]);
    }
  }

  // epilogue: tanh, packed b128 store to featT[n][m], column-sum partials
  float colsum[8] = {0.f, 0.f, 0.f, 0.f, 0.f, 0.f, 0.f, 0.f};
#pragma unroll
  for (int j = 0; j < 4; ++j) {
    v8bf pk;
#pragma unroll
    for (int r = 0; r < 8; ++r) {
      float v = fast_tanh(acc[j][r]);
      pk[r] = (__bf16)v;
      float s = v;
      s += __shfl_xor(s, 8, 16);
      s += __shfl_xor(s, 4, 16);
      s += __shfl_xor(s, 2, 16);
      s += __shfl_xor(s, 1, 16);
      colsum[r] += s;
    }
    int n = nblk + j * 16 + n16;
    *(v8bf*)(Yb + (size_t)n * M_ + m0 + half * 8) = pk;
  }
  if (n16 == 0) {
    float* dst = sumP + ((size_t)b * gridDim.x + blockIdx.x) * M_;
#pragma unroll
    for (int r = 0; r < 8; ++r) dst[m0 + half * 8 + r] = colsum[r];
  }
}

// ---------------- kernel 3: means from partials + the two tiny FCs ----------------
__global__ void k_smallfc(const float* __restrict__ qsumP, const float* __restrict__ isumP,
                          const float* __restrict__ Wqfc, const float* __restrict__ Wifc,
                          float* __restrict__ qfeatatt, float* __restrict__ ifeatatt) {
  __shared__ float imean[M_], qmean[M_];
  const int b = blockIdx.x;
  const int tid = threadIdx.x;
  if (tid < 128) {
    float s = 0.f;
    for (int p = 0; p < HW / 64; ++p) s += isumP[((size_t)b * (HW / 64) + p) * M_ + tid];
    imean[tid] = s * (1.0f / HW);
  } else {
    int m = tid - 128;
    float s = 0.f;
    for (int p = 0; p < T_ / 64; ++p) s += qsumP[((size_t)b * (T_ / 64) + p) * M_ + m];
    qmean[m] = s * (1.0f / T_);
  }
  __syncthreads();
  if (tid < 128) {
    const float* Wr = Wqfc + (size_t)tid * M_;
    float s = 0.f;
#pragma unroll 8
    for (int m = 0; m < M_; ++m) s += qmean[m] * Wr[m];
    qfeatatt[b * A_ + tid] = s;
  } else {
    int a = tid - 128;
    const float* Wr = Wifc + (size_t)a * M_;
    float s = 0.f;
#pragma unroll 8
    for (int m = 0; m < M_; ++m) s += imean[m] * Wr[m];
    ifeatatt[b * A_ + a] = s;
  }
}

// -------- kernel 4/5: logits = W_att . tanh(addv + W_proj @ featT) (no atomics) --------
__global__ void k_logits(const float* __restrict__ Wproj, const __bf16* __restrict__ featT,
                         int ncols, const float* __restrict__ addv,
                         const float* __restrict__ Watt, float* __restrict__ logits,
                         int col_off) {
  __shared__ __bf16 sA[M_ * STR];
  __shared__ float red[8 * 64];
  const int b = blockIdx.y;
  const int nblk = blockIdx.x * 64;
  const int tid = threadIdx.x;
  const int wave = tid >> 5, lane = tid & 31;
  const int n16 = lane & 15, half = lane >> 4;
  const int a0 = wave * 16;
  const __bf16* Xb = featT + (size_t)b * ncols * M_;
  v8f acc[4] = {};

  for (int k0 = 0; k0 < M_; k0 += 32) {
    __syncthreads();
#pragma unroll
    for (int it = 0; it < 4; ++it) {
      int s = tid * 4 + it * 1024;
      int m = s >> 5, kk = s & 31;
      float4 w = *(const float4*)(Wproj + (size_t)m * M_ + k0 + kk);
      v4bf p;
      p[0] = (__bf16)w.x; p[1] = (__bf16)w.y; p[2] = (__bf16)w.z; p[3] = (__bf16)w.w;
      *(v4bf*)(sA + m * STR + kk) = p;
    }
    __syncthreads();
    v16bf a = lds_load_a(sA + (a0 + n16) * STR, half);
#pragma unroll
    for (int j = 0; j < 4; ++j) {
      // B fragment straight from global: featT[n][m], k=m contiguous
      v16bf bm = glb_load_b(Xb + (size_t)(nblk + j * 16 + n16) * M_ + k0, half);
      acc[j] = wmma_bf16(a, bm, acc[j]);
    }
  }

#pragma unroll
  for (int j = 0; j < 4; ++j) {
    float s = 0.f;
#pragma unroll
    for (int r = 0; r < 8; ++r) {
      int aidx = a0 + half * 8 + r;
      float v = fast_tanh(addv[b * A_ + aidx] + acc[j][r]);
      s += Watt[aidx] * v;
    }
    s += __shfl_xor(s, 16, 32);  // combine halves (same n, disjoint a-rows)
    if (half == 0) red[wave * 64 + j * 16 + n16] = s;
  }
  __syncthreads();
  if (tid < 64) {
    float s = 0.f;
#pragma unroll
    for (int w = 0; w < 8; ++w) s += red[w * 64 + tid];
    logits[(size_t)b * NTOT + col_off + nblk + tid] = s;
  }
}

// ---------------- kernel 6: softmax over NTOT ----------------
__global__ void k_softmax(const float* __restrict__ logits, float* __restrict__ att) {
  const int b = blockIdx.x;
  const float* L = logits + (size_t)b * NTOT;
  float* O = att + (size_t)b * NTOT;
  __shared__ float red[256];
  const int tid = threadIdx.x;
  float mx = -3.402823466e38f;
  for (int i = tid; i < NTOT; i += 256) mx = fmaxf(mx, L[i]);
  red[tid] = mx; __syncthreads();
  for (int s = 128; s > 0; s >>= 1) {
    if (tid < s) red[tid] = fmaxf(red[tid], red[tid + s]);
    __syncthreads();
  }
  mx = red[0]; __syncthreads();
  float sum = 0.f;
  for (int i = tid; i < NTOT; i += 256) sum += __expf(L[i] - mx);
  red[tid] = sum; __syncthreads();
  for (int s = 128; s > 0; s >>= 1) {
    if (tid < s) red[tid] += red[tid + s];
    __syncthreads();
  }
  float inv = 1.0f / red[0];
  for (int i = tid; i < NTOT; i += 256) O[i] = __expf(L[i] - mx) * inv;
}

// ---------------- kernel 7: att_feat partials (coalesced over m) ----------------
__global__ void k_attpart(const __bf16* __restrict__ ifeatT, const __bf16* __restrict__ qfeatT,
                          const float* __restrict__ att, float* __restrict__ part) {
  const int gx = blockIdx.x;  // 0..7
  const int b = blockIdx.y;
  const int tid = threadIdx.x;
  const int m = tid & 127, sub = tid >> 7;
  const float* Ab = att + (size_t)b * NTOT;
  const __bf16* Fi = ifeatT + (size_t)b * HW * M_;
  float s = 0.f;
  int n0 = gx * 512 + sub * 256;
#pragma unroll 4
  for (int n = n0; n < n0 + 256; ++n) s += (float)Fi[(size_t)n * M_ + m] * Ab[n];
  if (gx == 7) {
    const __bf16* Fq = qfeatT + (size_t)b * T_ * M_;
    int t0 = sub * 64;
#pragma unroll 4
    for (int t = t0; t < t0 + 64; ++t) s += (float)Fq[(size_t)t * M_ + m] * Ab[HW + t];
  }
  __shared__ float lp[256];
  lp[tid] = s; __syncthreads();
  if (sub == 0) part[((size_t)b * 8 + gx) * M_ + m] = lp[m] + lp[m + 128];
}

__global__ void k_attcombine(const float* __restrict__ part, float* __restrict__ att_feat) {
  int idx = blockIdx.x * 256 + threadIdx.x;  // B*M = 8192
  int b = idx >> 7, m = idx & 127;
  float s = 0.f;
#pragma unroll
  for (int g = 0; g < 8; ++g) s += part[((size_t)b * 8 + g) * M_ + m];
  att_feat[idx] = s;
}

// ---------------- launcher ----------------
extern "C" void kernel_launch(void* const* d_in, const int* in_sizes, int n_in,
                              void* d_out, int out_size, void* d_ws, size_t ws_size,
                              hipStream_t stream) {
  (void)in_sizes; (void)n_in; (void)out_size; (void)ws_size;
  const float* img_feat = (const float*)d_in[0];
  const float* text_feat = (const float*)d_in[1];
  const float* W_img = (const float*)d_in[2];
  const float* W_txt = (const float*)d_in[3];
  const float* W_qfc = (const float*)d_in[4];
  const float* W_ifc = (const float*)d_in[5];
  const float* W_ipc = (const float*)d_in[6];
  const float* W_qpc = (const float*)d_in[7];
  const float* W_att = (const float*)d_in[8];

  char* ws = (char*)d_ws;
  size_t off = 0;
  __bf16* ifeatT = (__bf16*)(ws + off); off += (size_t)B_ * HW * M_ * 2;   // 64 MB
  __bf16* qfeatT = (__bf16*)(ws + off); off += (size_t)B_ * T_ * M_ * 2;   // 2 MB
  float* isumP   = (float*)(ws + off); off += (size_t)B_ * (HW / 64) * M_ * 4;  // 2 MB
  float* qsumP   = (float*)(ws + off); off += (size_t)B_ * (T_ / 64) * M_ * 4;
  float* logits  = (float*)(ws + off); off += (size_t)B_ * NTOT * 4;
  float* qfeatatt = (float*)(ws + off); off += (size_t)B_ * A_ * 4;
  float* ifeatatt = (float*)(ws + off); off += (size_t)B_ * A_ * 4;
  float* attpart  = (float*)(ws + off); off += (size_t)B_ * 8 * M_ * 4;

  float* out_attfeat = (float*)d_out;                // [B, M]
  float* out_att = (float*)d_out + (size_t)B_ * M_;  // [B, NTOT]

  k_ingemm<HID_, T_><<<dim3(T_ / 64, B_), dim3(256), 0, stream>>>(W_txt, text_feat,
                                                                  qfeatT, qsumP);
  k_ingemm<CIMG, HW><<<dim3(HW / 64, B_), dim3(256), 0, stream>>>(W_img, img_feat,
                                                                  ifeatT, isumP);
  k_smallfc<<<dim3(B_), dim3(256), 0, stream>>>(qsumP, isumP, W_qfc, W_ifc,
                                                qfeatatt, ifeatatt);
  k_logits<<<dim3(HW / 64, B_), dim3(256), 0, stream>>>(W_ipc, ifeatT, HW, qfeatatt,
                                                        W_att, logits, 0);
  k_logits<<<dim3(T_ / 64, B_), dim3(256), 0, stream>>>(W_qpc, qfeatT, T_, ifeatatt,
                                                        W_att, logits, HW);
  k_softmax<<<dim3(B_), dim3(256), 0, stream>>>(logits, out_att);
  k_attpart<<<dim3(8, B_), dim3(256), 0, stream>>>(ifeatT, qfeatT, out_att, attpart);
  k_attcombine<<<dim3(B_ * M_ / 256), dim3(256), 0, stream>>>(attpart, out_attfeat);
}
